// EarthSpecificBlock_25615184954004
// MI455X (gfx1250) — matrix operand order repaired and checked by reading, not compiled, not run on hardware
//
#include <hip/hip_runtime.h>
#include <hip/hip_bf16.h>

typedef __attribute__((ext_vector_type(16))) _Float16 v16h;
typedef __attribute__((ext_vector_type(8)))  float    v8f;
typedef __attribute__((ext_vector_type(4)))  int      v4i;

// ---------------- gfx1250 async global->LDS copy (ASYNCcnt) ----------------
#if defined(__has_builtin)
# if __has_builtin(__builtin_amdgcn_global_load_async_to_lds_b128) && \
     __has_builtin(__builtin_amdgcn_s_wait_asynccnt)
#  define HAVE_ASYNC_LDS 1
# endif
#endif
#ifndef HAVE_ASYNC_LDS
# define HAVE_ASYNC_LDS 0
#endif

#if HAVE_ASYNC_LDS
typedef __attribute__((address_space(1))) v4i* as1_v4i_p;
typedef __attribute__((address_space(3))) v4i* as3_v4i_p;
#endif

__device__ __forceinline__ void cp16(void* lds_dst, const void* gsrc) {
#if HAVE_ASYNC_LDS
  __builtin_amdgcn_global_load_async_to_lds_b128(
      (as1_v4i_p)(gsrc), (as3_v4i_p)(lds_dst), 0, 0);
#else
  *(uint4*)lds_dst = *(const uint4*)gsrc;
#endif
}
__device__ __forceinline__ void cp16_wait() {
#if HAVE_ASYNC_LDS
  __builtin_amdgcn_s_wait_asynccnt(0);
#endif
}

// ---------------- problem constants ----------------
static constexpr int L_TOK = 57600;      // Z*H*W = 8*60*120
static constexpr int CH    = 192;
static constexpr int NW    = 400;        // nZ*nH*nL = 4*10*10
static constexpr int NTOK  = 144;        // tokens per window
static constexpr int HEADS = 6;
static constexpr int HD    = 32;
static constexpr int QKVN  = 576;        // 3*C
static constexpr int FC1N  = 768;        // 4*C
static constexpr int TBL   = 3312;       // bias table rows

// ---------------- WMMA helpers (gfx1250 wave32 layouts, ISA 7.12.2) ----------------
__device__ __forceinline__ v8f wmma_f16(v16h a, v16h b, v8f c) {
  return __builtin_amdgcn_wmma_f32_16x16x32_f16(false, a, false, b, (short)0, c, false, false);
}

// A fragment 16x32 f16 from row-major matrix (ld in halfs), tile origin = base
__device__ __forceinline__ v16h load_frag_a(const _Float16* base, int ld) {
  int l  = threadIdx.x & 31;
  int m  = l & 15;
  int kb = (l >> 4) << 4;
  v16h a;
#pragma unroll
  for (int i = 0; i < 16; ++i) {
    int k = kb + i;
    a[i] = base[m * ld + k];
  }
  return a;
}

// B fragment 32x16 f16 from row-major KxN matrix (ld in halfs)
__device__ __forceinline__ v16h load_frag_b(const _Float16* base, int ld) {
  int l  = threadIdx.x & 31;
  int n  = l & 15;
  int kb = (l >> 4) << 4;
  v16h b;
#pragma unroll
  for (int i = 0; i < 16; ++i) b[i] = base[(kb + i) * ld + n];
  return b;
}

// B fragment where logical B(k,n) = base[n*ld + k] (M^T of an NxK row-major M)
__device__ __forceinline__ v16h load_frag_bt(const _Float16* base, int ld) {
  int l  = threadIdx.x & 31;
  int n  = l & 15;
  int kb = (l >> 4) << 4;
  v16h b;
#pragma unroll
  for (int i = 0; i < 16; ++i) b[i] = base[n * ld + kb + i];
  return b;
}

// ---------------- weight f32 -> f16 ----------------
__global__ void f32_to_f16_kernel(const float* __restrict__ src, _Float16* __restrict__ dst, int n) {
  int i = blockIdx.x * blockDim.x + threadIdx.x;
  if (i < n) dst[i] = (_Float16)src[i];
}

// ---------------- token -> (window, in-window) mapping (roll by -SS then partition) ----------
__device__ __forceinline__ void token_to_window(int t, int& w, int& n) {
  int z  = t / 7200;           // 60*120
  int h  = (t / 120) % 60;
  int wl = t % 120;
  int zr = (z + 7) & 7;        // roll -1 (mod 8)
  int hr = (h + 57) % 60;      // roll -3
  int lr = (wl + 114) % 120;   // roll -6
  int iz = zr >> 1, ih = hr / 6, il = lr / 12;
  int z1 = zr & 1,  h1 = hr % 6, w1 = lr % 12;
  w = (iz * 10 + ih) * 10 + il;
  n = (z1 * 6 + h1) * 12 + w1;
}

// ---------------- LN1 + modulate + window gather : one wave per token ----------------
__global__ __launch_bounds__(256)
void ln1_window_kernel(const float* __restrict__ x, const float* __restrict__ cond,
                       const float* __restrict__ g, const float* __restrict__ bt,
                       _Float16* __restrict__ hw) {
  int wid = threadIdx.x >> 5, lane = threadIdx.x & 31;
  int t = blockIdx.x * 8 + wid;
  const float* xp = x + (size_t)t * CH;
  float v[6], s = 0.f, s2 = 0.f;
#pragma unroll
  for (int j = 0; j < 6; ++j) { v[j] = xp[lane * 6 + j]; s += v[j]; s2 += v[j] * v[j]; }
#pragma unroll
  for (int m = 16; m; m >>= 1) { s += __shfl_xor(s, m, 32); s2 += __shfl_xor(s2, m, 32); }
  float mu = s * (1.f / CH);
  float rs = rsqrtf(s2 * (1.f / CH) - mu * mu + 1e-5f);
  int w, n; token_to_window(t, w, n);
  _Float16* op = hw + ((size_t)w * NTOK + n) * CH;
  const float* sh = cond;        // sh_msa
  const float* sc = cond + 192;  // sc_msa
#pragma unroll
  for (int j = 0; j < 6; ++j) {
    int c = lane * 6 + j;
    float u = (v[j] - mu) * rs * g[c] + bt[c];
    u = u * (1.f + sc[c]) + sh[c];
    op[c] = (_Float16)u;
  }
}

// ---------------- generic WMMA GEMM: C[M,N] = A[M,K](f16) @ B[K,N](f16) + bias ----
// EPI 0: store f16   EPI 1: store f32   EPI 2: GELU -> f16
// EPI 3: out_f32 = Y + gt[col]*(acc+bias)
template<int EPI>
__global__ __launch_bounds__(256)
void gemm_kernel(const _Float16* __restrict__ A, const _Float16* __restrict__ B,
                 const float* __restrict__ bias, int N, int K,
                 _Float16* __restrict__ o16, float* __restrict__ o32,
                 const float* __restrict__ Y, const float* __restrict__ gt) {
  __shared__ _Float16 As[128 * 32];
  __shared__ _Float16 Bs[32 * 64];
  int tid = threadIdx.x;
  int wid = tid >> 5;
  int wm = wid >> 1, wn = wid & 1;
  int row0 = blockIdx.y * 128;
  int col0 = blockIdx.x * 64;
  v8f acc[2][2];
#pragma unroll
  for (int i = 0; i < 2; ++i)
#pragma unroll
    for (int j = 0; j < 2; ++j) acc[i][j] = (v8f){};
  for (int k0 = 0; k0 < K; k0 += 32) {
#pragma unroll
    for (int r = 0; r < 2; ++r) {
      int vi = tid + 256 * r;
      int ar = vi >> 2, ac = (vi & 3) * 8;
      cp16(&As[ar * 32 + ac], &A[(size_t)(row0 + ar) * K + k0 + ac]);
    }
    {
      int br = tid >> 3, bc = (tid & 7) * 8;
      cp16(&Bs[br * 64 + bc], &B[(size_t)(k0 + br) * N + col0 + bc]);
    }
    if (k0 + 32 < K)  // hint next A tile toward cache (global_prefetch_b8)
      __builtin_prefetch(&A[(size_t)(row0 + (tid >> 1)) * K + k0 + 32], 0, 1);
    cp16_wait();
    __syncthreads();
#pragma unroll
    for (int j = 0; j < 2; ++j) {
      v16h b = load_frag_b(&Bs[wn * 32 + j * 16], 64);
#pragma unroll
      for (int i = 0; i < 2; ++i) {
        v16h a = load_frag_a(&As[(wm * 32 + i * 16) * 32], 32);
        acc[i][j] = wmma_f16(a, b, acc[i][j]);
      }
    }
    __syncthreads();
  }
  int l = tid & 31;
  int nn = l & 15, mg = (l >> 4) << 3;
#pragma unroll
  for (int i = 0; i < 2; ++i)
#pragma unroll
    for (int j = 0; j < 2; ++j)
#pragma unroll
      for (int v = 0; v < 8; ++v) {
        int gr = row0 + wm * 32 + i * 16 + mg + v;
        int gc = col0 + wn * 32 + j * 16 + nn;
        float val = acc[i][j][v] + bias[gc];
        size_t idx = (size_t)gr * N + gc;
        if (EPI == 0) o16[idx] = (_Float16)val;
        if (EPI == 1) o32[idx] = val;
        if (EPI == 2) {
          float gelu = 0.5f * val * (1.f + erff(val * 0.7071067811865475f));
          o16[idx] = (_Float16)gelu;
        }
        if (EPI == 3) o32[idx] = Y[idx] + gt[gc] * val;
      }
}

// ---------------- windowed attention: one block per (window, head) ----------------
__global__ __launch_bounds__(256)
void attn_kernel(const _Float16* __restrict__ qkv, const float* __restrict__ bias_table,
                 _Float16* __restrict__ out_win) {
  __shared__ _Float16 Ql[NTOK * HD];
  __shared__ _Float16 Kl[NTOK * HD];
  __shared__ _Float16 Vl[160 * HD];        // K padded 144 -> 160
  __shared__ _Float16 P [NTOK * 160];      // softmax probs, K-padded
  __shared__ float    biasS[TBL];          // bias slice for this (type, head)
  __shared__ int      qadd[NTOK];          // A(q) part of pos index
  __shared__ int      jadd[NTOK];          // B(j) part of pos index
  __shared__ int      regv[NTOK];          // shift-mask region id per token
  int blk = blockIdx.x;
  int w = blk / HEADS, head = blk % HEADS;
  int tid = threadIdx.x, wid = tid >> 5, lane = tid & 31;
  int typ = w / 10;                        // iz*nH + ih
  int iz = w / 100, ih = (w / 10) % 10;
  int toff = typ * HEADS + head;
  const float scale = 0.17677669529663688f;  // HD^-0.5
  const _Float16* qbase = qkv + (size_t)w * NTOK * QKVN + head * HD;

  // K, V via async global->LDS (16B chunks); Q staged with scale folded in
  for (int ch = tid; ch < NTOK * 4; ch += 256) {   // 4 x 16B per 32-half row
    int n = ch >> 2, c0 = (ch & 3) * 8;
    cp16(&Kl[n * HD + c0], &qbase[(size_t)n * QKVN + 192 + c0]);
    cp16(&Vl[n * HD + c0], &qbase[(size_t)n * QKVN + 384 + c0]);
  }
  for (int idx = tid; idx < NTOK * HD; idx += 256) {
    int n = idx >> 5, d = idx & 31;
    Ql[idx] = (_Float16)((float)qbase[(size_t)n * QKVN + d] * scale);
  }
  for (int idx = tid; idx < 16 * HD; idx += 256) Vl[NTOK * HD + idx] = (_Float16)0.f;  // V pad
  // per-token index tables: pidx(q,j) = qadd[q] + jadd[j] + 11
  for (int idx = tid; idx < NTOK; idx += 256) {
    int z = idx / 72, h = (idx / 12) % 6, wv = idx % 12;
    qadd[idx] = z * 828 + h * 23 + wv;
    jadd[idx] = 1656 * z + 138 * h - wv;
    int gz = iz * 2 + z, gh = ih * 6 + h;
    regv[idx] = (gz < 6 ? 0 : (gz == 6 ? 1 : 2)) * 3 + (gh < 54 ? 0 : (gh < 57 ? 1 : 2));
  }
  for (int idx = tid; idx < TBL; idx += 256)       // pre-gather bias slice
    biasS[idx] = bias_table[(size_t)idx * 240 + toff];
  cp16_wait();
  __syncthreads();

  // --- S = Q@K^T + bias + mask, softmax, write P ---
  for (int rt = wid; rt < 9; rt += 8) {
    v16h a = load_frag_a(&Ql[rt * 16 * HD], HD);
    int aq[8], rq[8];
#pragma unroll
    for (int v = 0; v < 8; ++v) {
      int q = rt * 16 + ((lane >> 4) << 3) + v;
      aq[v] = qadd[q] + 11;
      rq[v] = regv[q];
    }
    float s[9][8];
#pragma unroll
    for (int ct = 0; ct < 9; ++ct) {
      v16h b = load_frag_bt(&Kl[ct * 16 * HD], HD);
      v8f c = (v8f){};
      c = wmma_f16(a, b, c);
      int j  = ct * 16 + (lane & 15);
      int jb = jadd[j];
      int rk = regv[j];
#pragma unroll
      for (int v = 0; v < 8; ++v) {
        float bb = biasS[aq[v] + jb];
        s[ct][v] = c[v] + bb + ((rq[v] != rk) ? -100.f : 0.f);
      }
    }
#pragma unroll
    for (int v = 0; v < 8; ++v) {
      float mx = s[0][v];
#pragma unroll
      for (int ct = 1; ct < 9; ++ct) mx = fmaxf(mx, s[ct][v]);
#pragma unroll
      for (int m = 1; m < 16; m <<= 1) mx = fmaxf(mx, __shfl_xor(mx, m, 32));
      float p[9], sum = 0.f;
#pragma unroll
      for (int ct = 0; ct < 9; ++ct) { p[ct] = __expf(s[ct][v] - mx); sum += p[ct]; }
#pragma unroll
      for (int m = 1; m < 16; m <<= 1) sum += __shfl_xor(sum, m, 32);
      float inv = 1.f / sum;
      int q = rt * 16 + ((lane >> 4) << 3) + v;
#pragma unroll
      for (int ct = 0; ct < 9; ++ct)
        P[q * 160 + ct * 16 + (lane & 15)] = (_Float16)(p[ct] * inv);
    }
  }
  for (int idx = tid; idx < NTOK * 16; idx += 256) {   // zero K-pad cols 144..159
    int q = idx >> 4;
    P[q * 160 + 144 + (idx & 15)] = (_Float16)0.f;
  }
  __syncthreads();

  // --- O = P @ V ---
  for (int id = wid; id < 18; id += 8) {
    int rt = id >> 1, cn = id & 1;
    v8f acc = (v8f){};
#pragma unroll
    for (int ks = 0; ks < 5; ++ks) {
      v16h a = load_frag_a(&P[rt * 16 * 160 + ks * 32], 160);
      v16h b = load_frag_b(&Vl[ks * 32 * HD + cn * 16], HD);
      acc = wmma_f16(a, b, acc);
    }
    int d = cn * 16 + (lane & 15);
#pragma unroll
    for (int v = 0; v < 8; ++v) {
      int q = rt * 16 + ((lane >> 4) << 3) + v;
      out_win[((size_t)w * NTOK + q) * CH + head * HD + d] = (_Float16)acc[v];
    }
  }
}

// ---------------- scatter back + residual + LN2 + modulate : one wave per token ----------
__global__ __launch_bounds__(256)
void scatter_ln2_kernel(const float* __restrict__ x, const float* __restrict__ proj,
                        const float* __restrict__ cond, const float* __restrict__ g,
                        const float* __restrict__ bt, float* __restrict__ y,
                        _Float16* __restrict__ mlp_in) {
  int wid = threadIdx.x >> 5, lane = threadIdx.x & 31;
  int t = blockIdx.x * 8 + wid;
  int w, n; token_to_window(t, w, n);
  const float* xp = x + (size_t)t * CH;
  const float* pp = proj + ((size_t)w * NTOK + n) * CH;
  const float* gt_msa = cond + 384;
  float v[6], s = 0.f, s2 = 0.f;
#pragma unroll
  for (int j = 0; j < 6; ++j) {
    int c = lane * 6 + j;
    v[j] = xp[c] + gt_msa[c] * pp[c];
    s += v[j]; s2 += v[j] * v[j];
  }
#pragma unroll
  for (int m = 16; m; m >>= 1) { s += __shfl_xor(s, m, 32); s2 += __shfl_xor(s2, m, 32); }
  float mu = s * (1.f / CH);
  float rs = rsqrtf(s2 * (1.f / CH) - mu * mu + 1e-5f);
  float* yp = y + (size_t)t * CH;
  _Float16* mp = mlp_in + (size_t)t * CH;
  const float* sh = cond + 576;  // sh_mlp
  const float* sc = cond + 768;  // sc_mlp
#pragma unroll
  for (int j = 0; j < 6; ++j) {
    int c = lane * 6 + j;
    yp[c] = v[j];
    float u = (v[j] - mu) * rs * g[c] + bt[c];
    mp[c] = (_Float16)(u * (1.f + sc[c]) + sh[c]);
  }
}

// ---------------- host launcher ----------------
extern "C" void kernel_launch(void* const* d_in, const int* in_sizes, int n_in,
                              void* d_out, int out_size, void* d_ws, size_t ws_size,
                              hipStream_t stream) {
  const float* x          = (const float*)d_in[0];
  const float* cond       = (const float*)d_in[1];
  const float* g1         = (const float*)d_in[2];
  const float* bt1        = (const float*)d_in[3];
  const float* w_qkv      = (const float*)d_in[4];
  const float* b_qkv      = (const float*)d_in[5];
  const float* bias_table = (const float*)d_in[6];
  const float* w_proj     = (const float*)d_in[7];
  const float* b_proj     = (const float*)d_in[8];
  const float* g2         = (const float*)d_in[9];
  const float* bt2        = (const float*)d_in[10];
  const float* w_fc1      = (const float*)d_in[11];
  const float* b_fc1      = (const float*)d_in[12];
  const float* w_fc2      = (const float*)d_in[13];
  const float* b_fc2      = (const float*)d_in[14];
  float* out = (float*)d_out;

  char* ws = (char*)d_ws;
  size_t off = 0;
  auto carve = [&](size_t bytes) { size_t r = off; off += (bytes + 255) & ~(size_t)255; return r; };
  _Float16* wqkv_h  = (_Float16*)(ws + carve((size_t)CH * QKVN * 2));
  _Float16* wproj_h = (_Float16*)(ws + carve((size_t)CH * CH * 2));
  _Float16* wfc1_h  = (_Float16*)(ws + carve((size_t)CH * FC1N * 2));
  _Float16* wfc2_h  = (_Float16*)(ws + carve((size_t)FC1N * CH * 2));
  size_t hw_off   = carve((size_t)L_TOK * CH * 2);      // hw, later reused as out_win
  size_t qkv_off  = carve((size_t)L_TOK * QKVN * 2);    // qkv, later start of act
  size_t proj_off = carve((size_t)L_TOK * CH * 4);      // f32 proj out
  size_t y_off    = carve((size_t)L_TOK * CH * 4);      // f32 residual y
  size_t mlp_off  = carve((size_t)L_TOK * CH * 2);      // f16 mlp input
  _Float16* hw      = (_Float16*)(ws + hw_off);
  _Float16* qkv16   = (_Float16*)(ws + qkv_off);
  float*    projf   = (float*)(ws + proj_off);
  float*    yf      = (float*)(ws + y_off);
  _Float16* mlp_in  = (_Float16*)(ws + mlp_off);
  _Float16* out_win = hw;                                // reuse (hw dead after QKV gemm)
  _Float16* act16   = (_Float16*)(ws + qkv_off);         // reuse qkv+proj region (dead by fc1)

  f32_to_f16_kernel<<<(CH * QKVN + 255) / 256, 256, 0, stream>>>(w_qkv, wqkv_h, CH * QKVN);
  f32_to_f16_kernel<<<(CH * CH + 255) / 256, 256, 0, stream>>>(w_proj, wproj_h, CH * CH);
  f32_to_f16_kernel<<<(CH * FC1N + 255) / 256, 256, 0, stream>>>(w_fc1, wfc1_h, CH * FC1N);
  f32_to_f16_kernel<<<(FC1N * CH + 255) / 256, 256, 0, stream>>>(w_fc2, wfc2_h, FC1N * CH);

  ln1_window_kernel<<<L_TOK / 8, 256, 0, stream>>>(x, cond, g1, bt1, hw);

  gemm_kernel<0><<<dim3(QKVN / 64, L_TOK / 128), 256, 0, stream>>>(
      hw, wqkv_h, b_qkv, QKVN, CH, qkv16, nullptr, nullptr, nullptr);

  attn_kernel<<<NW * HEADS, 256, 0, stream>>>(qkv16, bias_table, out_win);

  gemm_kernel<1><<<dim3(CH / 64, L_TOK / 128), 256, 0, stream>>>(
      out_win, wproj_h, b_proj, CH, CH, nullptr, projf, nullptr, nullptr);

  scatter_ln2_kernel<<<L_TOK / 8, 256, 0, stream>>>(x, projf, cond, g2, bt2, yf, mlp_in);

  gemm_kernel<2><<<dim3(FC1N / 64, L_TOK / 128), 256, 0, stream>>>(
      mlp_in, wfc1_h, b_fc1, FC1N, CH, act16, nullptr, nullptr, nullptr);

  gemm_kernel<3><<<dim3(CH / 64, L_TOK / 128), 256, 0, stream>>>(
      act16, wfc2_h, b_fc2, CH, FC1N, nullptr, out, yf, cond + 960);
}
